// MLPBlock_2516850836030
// MI455X (gfx1250) — matrix operand order repaired and checked by reading, not compiled, not run on hardware
//
#include <hip/hip_runtime.h>
#include <stdint.h>

typedef unsigned short ushort_t;
typedef __attribute__((ext_vector_type(16))) __bf16 v16bf;
typedef __attribute__((ext_vector_type(8)))  float  v8f;
typedef __attribute__((ext_vector_type(2)))  __bf16 bf16x2;

#define TT   512
#define EE   16
#define KK   4
#define HH   2880
#define II   2880
#define R1   (2*II)        // 5760 rows of W1
#define R2   HH            // 2880 rows of W2
#define GH   (HH/32)       // 90
#define GI   (II/32)       // 90
#define CAP  512
#define ALPHA 1.702f
#define LIMIT 7.0f
#define EPSV  1e-5f
#define LDSTR 72           // LDS row stride (bf16 elems) for 64-wide K chunk, padded

// ---------------- workspace layout (bytes) ----------------
#define WS_TQ    ((size_t)0)                          // bf16 [T][H]
#define WS_TKE   (WS_TQ   + (size_t)TT*HH*2)          // int  [T][4]
#define WS_TKW   (WS_TKE  + (size_t)TT*4*4)           // f32  [T][4]
#define WS_CNT   (WS_TKW  + (size_t)TT*4*4)           // int  [E]
#define WS_TIDX  (WS_CNT  + (size_t)256)              // int  [E][CAP]
#define WS_TW    (WS_TIDX + (size_t)EE*CAP*4)         // f32  [E][CAP]
#define WS_HACT  (WS_TW   + (size_t)EE*CAP*4)         // bf16 [E][CAP][I]

__device__ __forceinline__ ushort_t f2bf(float f) {
    uint32_t u = __float_as_uint(f);
    uint32_t r = (u + 0x7FFFu + ((u >> 16) & 1u)) >> 16;
    return (ushort_t)r;
}

// FP4 (E2M1) nibble -> BF16, times 2^(scale-127), 8 values per source dword.
// Byte LUTs for magnitudes {0,.5,1,1.5,2,3,4,6} -> bf16 lo/hi bytes, sign = bit3.
__device__ __forceinline__ void dequant8(uint32_t w, uint32_t scale_pk, uint32_t out[4]) {
    const uint32_t A_LO = 0xC0804000u, B_LO = 0xC0800000u;   // lo bytes, entries 4-7 / 0-3
    const uint32_t A_HI = 0x40404040u, B_HI = 0x3F3F3F00u;   // hi bytes
    bf16x2 sv = __builtin_bit_cast(bf16x2, scale_pk);
#pragma unroll
    for (int h = 0; h < 2; ++h) {
        uint32_t ww  = h ? (w >> 16) : (w & 0xFFFFu);
        uint32_t t   = (ww & 0xFFu) | ((ww & 0xFF00u) << 8);
        uint32_t u   = (t | (t << 4)) & 0x0F0F0F0Fu;          // nibbles j -> byte j
        uint32_t sel = u & 0x07070707u;
        uint32_t sgn = (u & 0x08080808u) << 4;                // 0x80 in hi byte if negative
        uint32_t lo  = __builtin_amdgcn_perm(A_LO, B_LO, sel);
        uint32_t hi  = __builtin_amdgcn_perm(A_HI, B_HI, sel) | sgn;
        uint32_t p01 = __builtin_amdgcn_perm(hi, lo, 0x05010400u); // bf16(k0),bf16(k1)
        uint32_t p23 = __builtin_amdgcn_perm(hi, lo, 0x07030602u); // bf16(k2),bf16(k3)
        bf16x2 a = __builtin_bit_cast(bf16x2, p01); a = a * sv;    // exact pow2 scale
        bf16x2 b = __builtin_bit_cast(bf16x2, p23); b = b * sv;
        out[h * 2 + 0] = __builtin_bit_cast(uint32_t, a);
        out[h * 2 + 1] = __builtin_bit_cast(uint32_t, b);
    }
}

union ABu { v16bf v; uint4 q[2]; };

// ---------------- kernel 0: residual init (out = x) ----------------
__global__ __launch_bounds__(256) void k_init(const float4* __restrict__ x, float4* __restrict__ out, int n4) {
    int i = blockIdx.x * 256 + threadIdx.x;
    if (i < n4) out[i] = x[i];
}

// ---------------- kernel 1: RMSNorm + router(top4+softmax) ----------------
__global__ __launch_bounds__(256) void k_rms_router(const float* __restrict__ x,
                                                    const float* __restrict__ ns,
                                                    const float* __restrict__ gw,
                                                    const float* __restrict__ gb,
                                                    ushort_t* __restrict__ tq,
                                                    int* __restrict__ tk_e,
                                                    float* __restrict__ tk_w) {
    __shared__ float red[256];
    __shared__ float logits[EE];
    const int t = blockIdx.x, tid = threadIdx.x;
    const float* xr = x + (size_t)t * HH;

    float ss = 0.f;
    for (int h = tid; h < HH; h += 256) { float v = xr[h]; ss += v * v; }
    red[tid] = ss; __syncthreads();
    for (int s = 128; s > 0; s >>= 1) { if (tid < s) red[tid] += red[tid + s]; __syncthreads(); }
    const float rstd = rsqrtf(red[0] / (float)HH + EPSV);
    __syncthreads();

    float acc[EE];
#pragma unroll
    for (int e = 0; e < EE; ++e) acc[e] = 0.f;
    for (int h = tid; h < HH; h += 256) {
        float tv = xr[h] * rstd * ns[h];
        tq[(size_t)t * HH + h] = f2bf(tv);
#pragma unroll
        for (int e = 0; e < EE; ++e) acc[e] += tv * gw[(size_t)e * HH + h];
    }
#pragma unroll
    for (int e = 0; e < EE; ++e) {
        red[tid] = acc[e]; __syncthreads();
        for (int s = 128; s > 0; s >>= 1) { if (tid < s) red[tid] += red[tid + s]; __syncthreads(); }
        if (tid == 0) logits[e] = red[0] + gb[e];
        __syncthreads();
    }
    if (tid == 0) {
        float lv[EE];
#pragma unroll
        for (int e = 0; e < EE; ++e) lv[e] = logits[e];
        int   bi[KK]; float bv[KK];
#pragma unroll
        for (int j = 0; j < KK; ++j) {
            float best = -1e30f; int b = 0;
            for (int e = 0; e < EE; ++e) if (lv[e] > best) { best = lv[e]; b = e; }
            bv[j] = best; bi[j] = b; lv[b] = -1e30f;
        }
        float m = bv[0], s = 0.f, ex[KK];
#pragma unroll
        for (int j = 0; j < KK; ++j) { ex[j] = __expf(bv[j] - m); s += ex[j]; }
        float inv = 1.f / s;
#pragma unroll
        for (int j = 0; j < KK; ++j) { tk_e[t * KK + j] = bi[j]; tk_w[t * KK + j] = ex[j] * inv; }
    }
}

// ---------------- kernel 2: deterministic per-expert token gather ----------------
__global__ void k_gather(const int* __restrict__ tk_e, const float* __restrict__ tk_w,
                         int* __restrict__ cnt, int* __restrict__ tidx, float* __restrict__ tw) {
    int e = threadIdx.x;
    if (e >= EE) return;
    int c = 0;
    for (int t = 0; t < TT; ++t)
#pragma unroll
        for (int j = 0; j < KK; ++j)
            if (tk_e[t * KK + j] == e) { tidx[e * CAP + c] = t; tw[e * CAP + c] = tk_w[t * KK + j]; ++c; }
    cnt[e] = c;
}

// ---------------- kernel 3: MLP1 (WMMA bf16) + SwiGLU -> h_act ----------------
__global__ __launch_bounds__(256) void k_mlp1(const ushort_t* __restrict__ tq,
                                              const uint8_t* __restrict__ blk,
                                              const uint8_t* __restrict__ scl,
                                              const float*   __restrict__ bias,
                                              const int*     __restrict__ cnt,
                                              const int*     __restrict__ tidx,
                                              ushort_t*      __restrict__ hact) {
    const int e = blockIdx.z, nb = blockIdx.x * 64, mb = blockIdx.y * 128;
    const int c = cnt[e];
    if (mb >= c) return;

    __shared__ __align__(16) ushort_t ldsw[64 * LDSTR];
    __shared__ int stok[128];
    const int tid = threadIdx.x;
    if (tid < 128) { int s = mb + tid; stok[tid] = (s < c) ? tidx[e * CAP + s] : 0; }
    __syncthreads();

    const int lane = tid & 31, wv = tid >> 5, kh = lane >> 4, mrow = lane & 15;
    const ushort_t* arow = tq + (size_t)stok[wv * 16 + mrow] * HH;

    v8f acc[4];
#pragma unroll
    for (int nt = 0; nt < 4; ++nt) acc[nt] = (v8f)(0.f);

    for (int kc = 0; kc < HH / 64; ++kc) {
        const int k0 = kc * 64, kg = kc * 2;
#pragma unroll
        for (int i = 0; i < 2; ++i) {
            const int idx = tid + i * 256, r = idx >> 3, d = idx & 7;
            const size_t rg = (size_t)e * R1 + (nb + r);
            const uint8_t* bp = blk + (rg * GH + (size_t)(kg + (d >> 2))) * 16 + (d & 3) * 4;
            const uint32_t w  = *(const uint32_t*)bp;
            const uint32_t s  = scl[rg * GH + (size_t)(kg + (d >> 2))];
            const uint32_t sp = (s << 7) | (s << 23);
            uint32_t o[4]; dequant8(w, sp, o);
            uint4 q; q.x = o[0]; q.y = o[1]; q.z = o[2]; q.w = o[3];
            *(uint4*)&ldsw[r * LDSTR + d * 8] = q;
            if (kc + 1 < HH / 64) __builtin_prefetch(bp + 32, 0, 1);  // next k-chunk weights
        }
        __syncthreads();
#pragma unroll
        for (int kki = 0; kki < 2; ++kki) {
            const int kk = kki * 32;
            ABu a;
            const ushort_t* ap = arow + k0 + kk + kh * 8;
            a.q[0] = *(const uint4*)ap;
            a.q[1] = *(const uint4*)(ap + 16);
#pragma unroll
            for (int nt = 0; nt < 4; ++nt) {
                ABu b;
                const ushort_t* bp2 = &ldsw[(nt * 16 + mrow) * LDSTR + kk + kh * 8];
                b.q[0] = *(const uint4*)bp2;
                b.q[1] = *(const uint4*)(bp2 + 16);
                acc[nt] = __builtin_amdgcn_wmma_f32_16x16x32_bf16(false, a.v, false, b.v,
                                                                  (short)0, acc[nt], false, false);
            }
        }
        __syncthreads();
    }

    // epilogue: +bias, SwiGLU over (even,odd) weight-row pairs, store bf16
#pragma unroll
    for (int nt = 0; nt < 4; ++nt) {
        const int n = nb + nt * 16 + mrow;
        const float bn = bias[(size_t)e * R1 + n];
#pragma unroll
        for (int r = 0; r < 8; ++r) {
            float v = acc[nt][r] + bn;
            float p = __shfl_xor(v, 1);
            if ((mrow & 1) == 0) {
                float g = fminf(v, LIMIT);
                float l = fminf(fmaxf(p, -LIMIT), LIMIT);
                float hv = g * (1.f / (1.f + __expf(-ALPHA * g))) * (l + 1.f);
                int M = r + (kh << 3);
                int slot = mb + wv * 16 + M;
                if (slot < c) hact[((size_t)e * CAP + slot) * II + (n >> 1)] = f2bf(hv);
            }
        }
    }
}

// ---------------- kernel 4: MLP2 (WMMA bf16) + weighted accumulate ----------------
__global__ __launch_bounds__(256) void k_mlp2(const ushort_t* __restrict__ hact,
                                              const uint8_t* __restrict__ blk,
                                              const uint8_t* __restrict__ scl,
                                              const float*   __restrict__ bias,
                                              const int*     __restrict__ cnt,
                                              const int*     __restrict__ tidx,
                                              const float*   __restrict__ tw,
                                              float*         __restrict__ out) {
    const int e = blockIdx.z, nb = blockIdx.x * 64, mb = blockIdx.y * 128;
    const int c = cnt[e];
    if (mb >= c) return;

    __shared__ __align__(16) ushort_t ldsw[64 * LDSTR];
    __shared__ int   stok[128];
    __shared__ float stw[128];
    const int tid = threadIdx.x;
    if (tid < 128) {
        int s = mb + tid;
        stok[tid] = (s < c) ? tidx[e * CAP + s] : 0;
        stw[tid]  = (s < c) ? tw[e * CAP + s] : 0.f;
    }
    __syncthreads();

    const int lane = tid & 31, wv = tid >> 5, kh = lane >> 4, mrow = lane & 15;
    const ushort_t* arow = hact + ((size_t)e * CAP + (size_t)(mb + wv * 16 + mrow)) * II;

    v8f acc[4];
#pragma unroll
    for (int nt = 0; nt < 4; ++nt) acc[nt] = (v8f)(0.f);

    for (int kc = 0; kc < II / 64; ++kc) {
        const int k0 = kc * 64, kg = kc * 2;
#pragma unroll
        for (int i = 0; i < 2; ++i) {
            const int idx = tid + i * 256, r = idx >> 3, d = idx & 7;
            const size_t rg = (size_t)e * R2 + (nb + r);
            const uint8_t* bp = blk + (rg * GI + (size_t)(kg + (d >> 2))) * 16 + (d & 3) * 4;
            const uint32_t w  = *(const uint32_t*)bp;
            const uint32_t s  = scl[rg * GI + (size_t)(kg + (d >> 2))];
            const uint32_t sp = (s << 7) | (s << 23);
            uint32_t o[4]; dequant8(w, sp, o);
            uint4 q; q.x = o[0]; q.y = o[1]; q.z = o[2]; q.w = o[3];
            *(uint4*)&ldsw[r * LDSTR + d * 8] = q;
            if (kc + 1 < II / 64) __builtin_prefetch(bp + 32, 0, 1);
        }
        __syncthreads();
#pragma unroll
        for (int kki = 0; kki < 2; ++kki) {
            const int kk = kki * 32;
            ABu a;
            const ushort_t* ap = arow + k0 + kk + kh * 8;
            a.q[0] = *(const uint4*)ap;
            a.q[1] = *(const uint4*)(ap + 16);
#pragma unroll
            for (int nt = 0; nt < 4; ++nt) {
                ABu b;
                const ushort_t* bp2 = &ldsw[(nt * 16 + mrow) * LDSTR + kk + kh * 8];
                b.q[0] = *(const uint4*)bp2;
                b.q[1] = *(const uint4*)(bp2 + 16);
                acc[nt] = __builtin_amdgcn_wmma_f32_16x16x32_bf16(false, a.v, false, b.v,
                                                                  (short)0, acc[nt], false, false);
            }
        }
        __syncthreads();
    }

#pragma unroll
    for (int nt = 0; nt < 4; ++nt) {
        const int n = nb + nt * 16 + mrow;
        const float bn = bias[(size_t)e * HH + n];
#pragma unroll
        for (int r = 0; r < 8; ++r) {
            const int M = r + (kh << 3);
            const int sl = wv * 16 + M;          // slot - mb
            if (mb + sl < c) {
                float v = acc[nt][r] + bn;
                atomicAdd(&out[(size_t)stok[sl] * HH + n], stw[sl] * v);
            }
        }
    }
}

extern "C" void kernel_launch(void* const* d_in, const int* in_sizes, int n_in,
                              void* d_out, int out_size, void* d_ws, size_t ws_size,
                              hipStream_t stream) {
    (void)in_sizes; (void)n_in; (void)out_size; (void)ws_size;
    const float*   x      = (const float*)d_in[0];
    const float*   ns     = (const float*)d_in[1];
    const float*   gw     = (const float*)d_in[2];
    const float*   gb     = (const float*)d_in[3];
    const uint8_t* m1b    = (const uint8_t*)d_in[4];
    const uint8_t* m1s    = (const uint8_t*)d_in[5];
    const float*   m1bias = (const float*)d_in[6];
    const uint8_t* m2b    = (const uint8_t*)d_in[7];
    const uint8_t* m2s    = (const uint8_t*)d_in[8];
    const float*   m2bias = (const float*)d_in[9];
    float* out = (float*)d_out;

    char* ws = (char*)d_ws;
    ushort_t* tq   = (ushort_t*)(ws + WS_TQ);
    int*      tk_e = (int*)     (ws + WS_TKE);
    float*    tk_w = (float*)   (ws + WS_TKW);
    int*      cnt  = (int*)     (ws + WS_CNT);
    int*      tidx = (int*)     (ws + WS_TIDX);
    float*    twp  = (float*)   (ws + WS_TW);
    ushort_t* hact = (ushort_t*)(ws + WS_HACT);

    const int n4 = (TT * HH) / 4;
    k_init<<<(n4 + 255) / 256, 256, 0, stream>>>((const float4*)x, (float4*)out, n4);
    k_rms_router<<<TT, 256, 0, stream>>>(x, ns, gw, gb, tq, tk_e, tk_w);
    k_gather<<<1, 32, 0, stream>>>(tk_e, tk_w, cnt, tidx, twp);
    k_mlp1<<<dim3(R1 / 64, CAP / 128, EE), 256, 0, stream>>>(tq, m1b, m1s, m1bias, cnt, tidx, hact);
    k_mlp2<<<dim3(HH / 64, CAP / 128, EE), 256, 0, stream>>>(hact, m2b, m2s, m2bias, cnt, tidx, twp, out);
}
